// RelPosMultiHeadAttention_54254026883397
// MI455X (gfx1250) — matrix operand order repaired and checked by reading, not compiled, not run on hardware
//
#include <hip/hip_runtime.h>
#include <hip/hip_bf16.h>

// ---------------------------------------------------------------------------
// Transformer-XL rel-pos multi-head attention for MI455X (gfx1250, wave32).
// bf16 WMMA (v_wmma_f32_16x16x32_bf16) everywhere; f32 accumulate.
// ---------------------------------------------------------------------------

#define EMBED  1024
#define HEADS  16
#define HDIM   64
#define BSZ    4
#define SEQ    1024
#define SCALE  0.125f   // 1/sqrt(64)

typedef __bf16 bf16_t;
typedef bf16_t v16bf __attribute__((ext_vector_type(16)));
typedef bf16_t v2bf  __attribute__((ext_vector_type(2)));
typedef float  v8f   __attribute__((ext_vector_type(8)));

union Frag { uint4 u[2]; v16bf v; };

// fp32 -> bf16 via native converts (RNE), packed pairwise so the backend can
// emit v_cvt_pk_bf16_f32 instead of a manual round/shift VALU chain.
__device__ __forceinline__ unsigned short f2bf(float f) {
  return __builtin_bit_cast(unsigned short, (bf16_t)f);
}
__device__ __forceinline__ unsigned pack2_bf16(float x, float y) {
#if __has_builtin(__builtin_amdgcn_cvt_pk_bf16_f32)
  return __builtin_bit_cast(unsigned, __builtin_amdgcn_cvt_pk_bf16_f32(x, y));
#else
  v2bf p; p[0] = (bf16_t)x; p[1] = (bf16_t)y;
  return __builtin_bit_cast(unsigned, p);
#endif
}

__device__ __forceinline__ v8f wmma_bf16(const Frag& a, const Frag& b, v8f c) {
  return __builtin_amdgcn_wmma_f32_16x16x32_bf16(
      false, a.v, false, b.v, (short)0, c, false, false);
}

// ---------------------------------------------------------------------------
// Projection GEMM: C[M,1024] = A[M,1024] @ W[1024,1024]^T  (C[m,n]=sum A[m,k]W[n,k])
// 128x128 tile, 8 waves (2x4), K-step 32.  Epilogue modes:
//   0: q  -> two bf16 outputs (+r_w_bias / +r_r_bias), layout [b][h][i][d]
//   1: k/v-> one bf16 output,                      layout [b][h][j][d]
//   2: rr -> one bf16 output,                      layout [h][j][d]
//   3: attn_out -> f32 row-major (A is bf16 attn_vec)
// ---------------------------------------------------------------------------
__global__ __launch_bounds__(256) void proj_gemm(
    const float* __restrict__ A32, const unsigned short* __restrict__ A16,
    const float* __restrict__ W,
    const float* __restrict__ bias1, const float* __restrict__ bias2,
    unsigned short* __restrict__ out1, unsigned short* __restrict__ out2,
    float* __restrict__ outF, int mode)
{
  __shared__ unsigned short sA[128 * 40];   // bf16 tile, stride 40 (80B = 5*16B)
  __shared__ unsigned short sB[128 * 40];

  const int tid  = threadIdx.x;
  const int lane = tid & 31, wave = tid >> 5;
  const int hi   = lane >> 4, ln = lane & 15;
  const int mBase = blockIdx.y * 128;
  const int nBase = blockIdx.x * 128;
  const int wm = wave & 1;          // 2 wave-rows  (64 M each)
  const int wn = wave >> 1;         // 4 wave-cols  (32 N each)

  v8f acc[4][2];
  v8f zero = {0.f,0.f,0.f,0.f,0.f,0.f,0.f,0.f};
  for (int a = 0; a < 4; ++a) for (int b = 0; b < 2; ++b) acc[a][b] = zero;

  const int rowA = tid >> 3;                // this thread's staging row
  const int c4   = (tid & 7) * 4;           // and 4-float column offset

  for (int kc = 0; kc < EMBED; kc += 32) {
    // ---- stage A tile 128x32 (convert fp32 -> bf16 unless mode 3) ----
    if (mode == 3) {
      for (int it = 0; it < 4; ++it) {
        int f = tid + it * 256; int row = f >> 3; int cc = (f & 7) * 4;
        uint2 d = *(const uint2*)(A16 + (size_t)(mBase + row) * EMBED + kc + cc);
        *(uint2*)&sA[row * 40 + cc] = d;
      }
    } else {
      for (int it = 0; it < 4; ++it) {
        int f = tid + it * 256; int row = f >> 3; int cc = (f & 7) * 4;
        float4 d = *(const float4*)(A32 + (size_t)(mBase + row) * EMBED + kc + cc);
        uint2 p;
        p.x = pack2_bf16(d.x, d.y);
        p.y = pack2_bf16(d.z, d.w);
        *(uint2*)&sA[row * 40 + cc] = p;
      }
    }
    // ---- stage W tile 128x32 ----
    for (int it = 0; it < 4; ++it) {
      int f = tid + it * 256; int row = f >> 3; int cc = (f & 7) * 4;
      float4 d = *(const float4*)(W + (size_t)(nBase + row) * EMBED + kc + cc);
      uint2 p;
      p.x = pack2_bf16(d.x, d.y);
      p.y = pack2_bf16(d.z, d.w);
      *(uint2*)&sB[row * 40 + cc] = p;
    }
    // warm next K-chunk into cache while WMMAs run (global_prefetch_b8)
    if (kc + 32 < EMBED) {
      if (mode != 3)
        __builtin_prefetch(A32 + (size_t)(mBase + rowA) * EMBED + kc + 32 + c4, 0, 1);
      __builtin_prefetch(W + (size_t)(nBase + rowA) * EMBED + kc + 32 + c4, 0, 1);
    }
    __syncthreads();

    Frag af[4], bfr[2];
    for (int mt = 0; mt < 4; ++mt) {
      int r = wm * 64 + mt * 16 + ln;
      af[mt].u[0] = *(const uint4*)&sA[r * 40 + hi * 8];
      af[mt].u[1] = *(const uint4*)&sA[r * 40 + 16 + hi * 8];
    }
    for (int nt = 0; nt < 2; ++nt) {
      int r = wn * 32 + nt * 16 + ln;
      bfr[nt].u[0] = *(const uint4*)&sB[r * 40 + hi * 8];
      bfr[nt].u[1] = *(const uint4*)&sB[r * 40 + 16 + hi * 8];
    }
    for (int mt = 0; mt < 4; ++mt)
      for (int nt = 0; nt < 2; ++nt)
        acc[mt][nt] = wmma_bf16(af[mt], bfr[nt], acc[mt][nt]);
    __syncthreads();
  }

  // ---- epilogue ----
  for (int mt = 0; mt < 4; ++mt) {
    for (int nt = 0; nt < 2; ++nt) {
      int nOut = nBase + wn * 32 + nt * 16 + ln;
      for (int r = 0; r < 8; ++r) {
        int m = mBase + wm * 64 + mt * 16 + hi * 8 + r;
        float v = acc[mt][nt][r];
        if (mode == 0) {
          int i = m >> 2, b = m & 3;
          int head = nOut >> 6, d = nOut & 63;
          size_t idx = ((size_t)(b * HEADS + head) * SEQ + i) * HDIM + d;
          out1[idx] = f2bf(v + bias1[nOut]);
          out2[idx] = f2bf(v + bias2[nOut]);
        } else if (mode == 1) {
          int j = m >> 2, b = m & 3;
          int head = nOut >> 6, d = nOut & 63;
          out1[((size_t)(b * HEADS + head) * SEQ + j) * HDIM + d] = f2bf(v);
        } else if (mode == 2) {
          int head = nOut >> 6, d = nOut & 63;
          out1[((size_t)head * SEQ + m) * HDIM + d] = f2bf(v);
        } else {
          outF[(size_t)m * EMBED + nOut] = v;
        }
      }
    }
  }
}

// ---------------------------------------------------------------------------
// Attention: one block per (b,head, 32-query-row tile).  Scores stay in LDS.
//   Phase 1: S = qw @ k^T                    (WMMA, 8 j-chunks of 128)
//   Phase 2: S += rel_shift(qr @ rr^T)       (WMMA + exact-shift scatter-add)
//   Phase 3: row softmax reductions
//   Phase 4: prob -> global + bf16 LDS;  attn_vec += P @ V (WMMA)
// Dynamic LDS partition (bytes):
//   sS  f32[32*1032]      @      0   (132096)
//   sK  u16[128*72]       @ 132096   (18432)  (reused as sP  u16[32*136])
//   sR  u16[128*72]       @ 150528   (18432)  (reused as sVt u16[64*136])
//   sQw u16[32*72]        @ 168960   ( 4608)
//   sQr u16[48*72]        @ 173568   ( 6912)
//   sMax f32[32]          @ 180480
//   sInv f32[32]          @ 180608
//   sRed f32[32*8]        @ 180736    -> total 181760 B (CDNA5: 320KB/WG ok)
// ---------------------------------------------------------------------------
#define SSTR 1032

__global__ __launch_bounds__(256) void attn_kernel(
    const unsigned short* __restrict__ qw, const unsigned short* __restrict__ qr,
    const unsigned short* __restrict__ kb, const unsigned short* __restrict__ vb,
    const unsigned short* __restrict__ rrb,
    float* __restrict__ probOut, unsigned short* __restrict__ av)
{
  extern __shared__ __align__(16) char smem[];
  float*          sS   = (float*)smem;
  unsigned short* sK   = (unsigned short*)(smem + 132096);
  unsigned short* sR   = (unsigned short*)(smem + 150528);
  unsigned short* sQw  = (unsigned short*)(smem + 168960);
  unsigned short* sQr  = (unsigned short*)(smem + 173568);
  float*          sMax = (float*)(smem + 180480);
  float*          sInv = (float*)(smem + 180608);
  float*          sRed = (float*)(smem + 180736);
  unsigned short* sP   = sK;   // reuse in phase 4
  unsigned short* sVt  = sR;   // reuse in phase 4

  const int tid  = threadIdx.x;
  const int lane = tid & 31, wave = tid >> 5;
  const int hi   = lane >> 4, ln = lane & 15;
  const int i0   = blockIdx.x * 32;
  const int bn   = blockIdx.y;             // b*16 + head
  const int b    = bn >> 4, n = bn & 15;
  const size_t bnOff = (size_t)bn * SEQ * HDIM;
  const size_t nOff  = (size_t)n  * SEQ * HDIM;
  v8f zero = {0.f,0.f,0.f,0.f,0.f,0.f,0.f,0.f};

  // ---- load q tiles (qw: 32 rows; qr: 48 rows, zero-pad past seq end) ----
  { int r = tid >> 3, d0 = (tid & 7) * 8;
    *(uint4*)&sQw[r * 72 + d0] =
        *(const uint4*)(qw + bnOff + (size_t)(i0 + r) * HDIM + d0); }
  for (int f = tid; f < 48 * 8; f += 256) {
    int r = f >> 3, d0 = (f & 7) * 8;
    int gi = i0 + r;
    uint4 d = make_uint4(0u, 0u, 0u, 0u);
    if (gi < SEQ) d = *(const uint4*)(qr + bnOff + (size_t)gi * HDIM + d0);
    *(uint4*)&sQr[r * 72 + d0] = d;
  }

  // ================= Phase 1: AC = qw @ k^T =================
  for (int jc = 0; jc < 8; ++jc) {
    __syncthreads();
    for (int it = 0; it < 4; ++it) {
      int f = tid + it * 256; int j = f >> 3, d0 = (f & 7) * 8;
      *(uint4*)&sK[j * 72 + d0] =
          *(const uint4*)(kb + bnOff + (size_t)(jc * 128 + j) * HDIM + d0);
    }
    __syncthreads();
    v8f acc[2]; acc[0] = zero; acc[1] = zero;
    for (int ks = 0; ks < 2; ++ks) {
      Frag bq; int jrow = wave * 16 + ln;
      bq.u[0] = *(const uint4*)&sK[jrow * 72 + ks * 32 + hi * 8];
      bq.u[1] = *(const uint4*)&sK[jrow * 72 + ks * 32 + hi * 8 + 16];
      for (int mt = 0; mt < 2; ++mt) {
        Frag aq; int r = mt * 16 + ln;
        aq.u[0] = *(const uint4*)&sQw[r * 72 + ks * 32 + hi * 8];
        aq.u[1] = *(const uint4*)&sQw[r * 72 + ks * 32 + hi * 8 + 16];
        acc[mt] = wmma_bf16(aq, bq, acc[mt]);
      }
    }
    for (int mt = 0; mt < 2; ++mt)
      for (int r8 = 0; r8 < 8; ++r8) {
        int rl = mt * 16 + hi * 8 + r8;
        int j  = jc * 128 + wave * 16 + ln;
        sS[rl * SSTR + j] = acc[mt][r8];
      }
  }

  // ========== Phase 2: BDraw = qr @ rr^T, scatter-add with exact rel_shift ==
  // out[i,j]: j<=i  <- BDraw[i,   j-i+L-1]
  //           j==i+1<- 0
  //           j>i+1 <- BDraw[i+1, j-i-2]      (TXL wrap, no masking in ref)
  __syncthreads();
  for (int jc = 0; jc < 8; ++jc) {
    for (int it = 0; it < 4; ++it) {
      int f = tid + it * 256; int j = f >> 3, d0 = (f & 7) * 8;
      *(uint4*)&sR[j * 72 + d0] =
          *(const uint4*)(rrb + nOff + (size_t)(jc * 128 + j) * HDIM + d0);
    }
    __syncthreads();
    v8f acc[3]; acc[0] = zero; acc[1] = zero; acc[2] = zero;
    for (int ks = 0; ks < 2; ++ks) {
      Frag br; int jrow = wave * 16 + ln;
      br.u[0] = *(const uint4*)&sR[jrow * 72 + ks * 32 + hi * 8];
      br.u[1] = *(const uint4*)&sR[jrow * 72 + ks * 32 + hi * 8 + 16];
      for (int mt = 0; mt < 3; ++mt) {
        Frag aq; int r = mt * 16 + ln;
        aq.u[0] = *(const uint4*)&sQr[r * 72 + ks * 32 + hi * 8];
        aq.u[1] = *(const uint4*)&sQr[r * 72 + ks * 32 + hi * 8 + 16];
        acc[mt] = wmma_bf16(aq, br, acc[mt]);
      }
    }
    for (int mt = 0; mt < 3; ++mt)
      for (int r8 = 0; r8 < 8; ++r8) {
        int r = i0 + mt * 16 + hi * 8 + r8;        // absolute BDraw row
        int c = jc * 128 + wave * 16 + ln;         // absolute BDraw col
        if (r < SEQ) {
          int idst, jdst;
          if (c >= SEQ - 1 - r) { idst = r;     jdst = c + r - (SEQ - 1); }
          else                  { idst = r - 1; jdst = c + r + 1; }
          int rl = idst - i0;
          if (rl >= 0 && rl < 32)
            sS[rl * SSTR + jdst] += acc[mt][r8];   // injective map: no races
        }
      }
    __syncthreads();
  }

  // ================= Phase 3: softmax reductions =================
  {
    int r = tid >> 3, q8 = tid & 7;
    const float* row = &sS[r * SSTR];
    float mx = -3.4e38f;
    for (int t = 0; t < 128; ++t) mx = fmaxf(mx, row[q8 + 8 * t]);
    sRed[r * 8 + q8] = mx;
    __syncthreads();
    float m = sRed[r * 8];
    for (int t = 1; t < 8; ++t) m = fmaxf(m, sRed[r * 8 + t]);
    __syncthreads();
    float sum = 0.f;
    for (int t = 0; t < 128; ++t)
      sum += __expf((row[q8 + 8 * t] - m) * SCALE);
    sRed[r * 8 + q8] = sum;
    __syncthreads();
    if (q8 == 0) {
      float s = 0.f;
      for (int t = 0; t < 8; ++t) s += sRed[r * 8 + t];
      sMax[r] = m;
      sInv[r] = 1.0f / s;
    }
  }
  __syncthreads();

  // ========== Phase 4: write probs, attn_vec += P @ V ==========
  const int wm2 = wave & 1;          // 2 x 16 query rows
  const int wn2 = wave >> 1;         // 4 x 16 head dims
  v8f vacc = zero;
  for (int jc = 0; jc < 8; ++jc) {
    __syncthreads();
    for (int f = tid; f < 32 * 128; f += 256) {
      int rl = f >> 7, jl = f & 127;
      int j  = jc * 128 + jl;
      float p = __expf((sS[rl * SSTR + j] - sMax[rl]) * SCALE) * sInv[rl];
      int i = i0 + rl;
      probOut[(((size_t)i * SEQ + j) * BSZ + b) * HEADS + n] = p;
      sP[rl * 136 + jl] = f2bf(p);
    }
    for (int it = 0; it < 4; ++it) {   // V chunk, transposed into sVt[d][j]
      int f = tid + it * 256; int j = f >> 3, d0 = (f & 7) * 8;
      uint4 d = *(const uint4*)(vb + bnOff + (size_t)(jc * 128 + j) * HDIM + d0);
      const unsigned short* e = (const unsigned short*)&d;
      for (int q = 0; q < 8; ++q) sVt[(d0 + q) * 136 + j] = e[q];
    }
    __syncthreads();
    for (int ks = 0; ks < 4; ++ks) {
      Frag ap, bp;
      int r = wm2 * 16 + ln;
      ap.u[0] = *(const uint4*)&sP[r * 136 + ks * 32 + hi * 8];
      ap.u[1] = *(const uint4*)&sP[r * 136 + ks * 32 + hi * 8 + 16];
      int dr = wn2 * 16 + ln;
      bp.u[0] = *(const uint4*)&sVt[dr * 136 + ks * 32 + hi * 8];
      bp.u[1] = *(const uint4*)&sVt[dr * 136 + ks * 32 + hi * 8 + 16];
      vacc = wmma_bf16(ap, bp, vacc);
    }
  }
  { // attn_vec (bf16) in [i*BSZ+b][head*64+d] layout for the Wo GEMM
    int d = wn2 * 16 + ln;
    for (int r8 = 0; r8 < 8; ++r8) {
      int i = i0 + wm2 * 16 + hi * 8 + r8;
      av[((size_t)i * BSZ + b) * EMBED + n * HDIM + d] = f2bf(vacc[r8]);
    }
  }
}

// ---------------------------------------------------------------------------
// Host side: inputs per setup_inputs() order.  Workspace (bf16): 42 MB total.
// ---------------------------------------------------------------------------
extern "C" void kernel_launch(void* const* d_in, const int* in_sizes, int n_in,
                              void* d_out, int out_size, void* d_ws, size_t ws_size,
                              hipStream_t stream) {
  (void)in_sizes; (void)n_in; (void)out_size; (void)ws_size;
  const float* query    = (const float*)d_in[0];
  const float* key      = (const float*)d_in[1];
  const float* value    = (const float*)d_in[2];
  const float* r        = (const float*)d_in[3];
  const float* Wq       = (const float*)d_in[4];
  const float* Wk       = (const float*)d_in[5];
  const float* Wv       = (const float*)d_in[6];
  const float* Wr       = (const float*)d_in[7];
  const float* Wo       = (const float*)d_in[8];
  const float* r_w_bias = (const float*)d_in[9];
  const float* r_r_bias = (const float*)d_in[10];

  char* ws = (char*)d_ws;
  unsigned short* qw  = (unsigned short*)(ws);             //  8 MB
  unsigned short* qr  = (unsigned short*)(ws +  8388608);  //  8 MB
  unsigned short* kb  = (unsigned short*)(ws + 16777216);  //  8 MB
  unsigned short* vb  = (unsigned short*)(ws + 25165824);  //  8 MB
  unsigned short* rrb = (unsigned short*)(ws + 33554432);  //  2 MB
  unsigned short* av  = (unsigned short*)(ws + 35651584);  //  8 MB

  float* attn_out = (float*)d_out;                         // [1024,4,1024]
  float* probOut  = (float*)d_out + 4194304;               // [1024,1024,4,16]

  dim3 blk(256);
  proj_gemm<<<dim3(8, 32), blk, 0, stream>>>(query, nullptr, Wq, r_w_bias,
      r_r_bias, qw, qr, nullptr, 0);
  proj_gemm<<<dim3(8, 32), blk, 0, stream>>>(key, nullptr, Wk, nullptr,
      nullptr, kb, nullptr, nullptr, 1);
  proj_gemm<<<dim3(8, 32), blk, 0, stream>>>(value, nullptr, Wv, nullptr,
      nullptr, vb, nullptr, nullptr, 1);
  proj_gemm<<<dim3(8, 8),  blk, 0, stream>>>(r, nullptr, Wr, nullptr,
      nullptr, rrb, nullptr, nullptr, 2);
  attn_kernel<<<dim3(32, 64), blk, 181760, stream>>>(qw, qr, kb, vb, rrb,
      probOut, av);
  proj_gemm<<<dim3(8, 32), blk, 0, stream>>>(nullptr, av, Wo, nullptr,
      nullptr, nullptr, nullptr, attn_out, 3);
}
